// BreakthroughSNN_17325898072591
// MI455X (gfx1250) — matrix operand order repaired and checked by reading, not compiled, not run on hardware
//
#include <hip/hip_runtime.h>
#include <hip/hip_bf16.h>

// ---------------------------------------------------------------------------
// BreakthroughSNN for MI455X (gfx1250): bf16 WMMA everywhere, fp32 accumulate.
// ---------------------------------------------------------------------------

#define B_ 16
#define S_ 256
#define V_ 32000
#define D_ 512
#define L_ 4
#define DECAY_ 0.6065306597126334f   // exp(-1/2)
#define LN_EPS_ 1e-5f
#define LDA_ 520                     // padded LDS stride (halfs), 16B-aligned rows
#define LDA2_ 136                    // logits kernel chunk stride (halfs)

typedef __attribute__((ext_vector_type(16))) __bf16 v16bf;
typedef __attribute__((ext_vector_type(8)))  float  v8f;

union FragB16 { uint4 q[2]; v16bf v; };

static __device__ __forceinline__ unsigned short f2bf(float f) {
    unsigned int u = __float_as_uint(f);
    unsigned int r = u + 0x7FFFu + ((u >> 16) & 1u);   // round-to-nearest-even
    return (unsigned short)(r >> 16);
}

// Prefetch a 512KB bf16 weight matrix: 1024 threads x 512B stride.
static __device__ __forceinline__ void prefetch_w(const unsigned short* W, int tid) {
    __builtin_prefetch(W + (size_t)tid * 256, 0, 1);   // -> global_prefetch_b8
}

// Wave computes one 16x16 output tile at columns [n0, n0+16).
// A: 16xK bf16 in LDS (row-major, stride LDA_). B: W row-major [N][K] bf16 in
// global memory, used as K x 16 tile via transpose addressing (B[k][n]=W[n][k]).
static __device__ __forceinline__ v8f matmul16(const unsigned short* sA,
                                               const unsigned short* W,
                                               int n0, int lane) {
    v8f acc = {0.f, 0.f, 0.f, 0.f, 0.f, 0.f, 0.f, 0.f};
    const unsigned short* arow = sA + (lane & 15) * LDA_ + ((lane >> 4) * 8);
    const unsigned short* brow = W + (size_t)(n0 + (lane & 15)) * D_ + ((lane >> 4) * 8);
#pragma unroll 4                         // cap in-flight fragments: no spills
    for (int k0 = 0; k0 < D_; k0 += 32) {
        FragB16 A, Bf;
        A.q[0]  = *(const uint4*)(arow + k0);        // K = kb .. kb+7
        A.q[1]  = *(const uint4*)(arow + k0 + 16);   // K = kb+16 .. kb+23
        Bf.q[0] = *(const uint4*)(brow + k0);        // K = kb .. kb+7
        Bf.q[1] = *(const uint4*)(brow + k0 + 16);   // K = kb+16 .. kb+23
        acc = __builtin_amdgcn_wmma_f32_16x16x32_bf16(false, A.v, false, Bf.v,
                                                      (short)0, acc, false, false);
    }
    return acc;
}

// LIF step on a wave's 16x16 fragment. mem layout: [16][512] f32 in workspace.
static __device__ __forceinline__ void lif_frag(const v8f z, const float* bias,
                                                float* mem, int n0, int lane,
                                                float* spike, float* memv) {
    int n  = n0 + (lane & 15);
    int mb = (lane >> 4) * 8;
    float bn = bias[n];
#pragma unroll
    for (int i = 0; i < 8; i++) {
        int idx = (mb + i) * D_ + n;
        float m_ = mem[idx] * DECAY_ + z[i] + bn;
        float s  = (m_ - 1.0f >= 0.f) ? 1.f : 0.f;
        m_ *= (1.f - s);
        mem[idx] = m_;
        spike[i] = s;
        memv[i]  = m_;
    }
}

static __device__ __forceinline__ float wave_sum(float s) {
#pragma unroll
    for (int off = 16; off > 0; off >>= 1) s += __shfl_xor(s, off, 32);
    return s;
}

// Row LayerNorm over D=512: values held as fragments (8 per thread).
static __device__ __forceinline__ void layer_norm_frag(const float* val, float* outv,
                                                       const float* g, const float* b,
                                                       float* sLN, float* sStats,
                                                       int lane, int wave, int n0) {
    int mb = (lane >> 4) * 8;
    int n  = n0 + (lane & 15);
    __syncthreads();
#pragma unroll
    for (int i = 0; i < 8; i++) sLN[(mb + i) * D_ + n] = val[i];
    __syncthreads();
    if (wave < 16) {
        const float* row = sLN + wave * D_;
        float s = 0.f, s2 = 0.f;
        for (int c = lane; c < D_; c += 32) { float v = row[c]; s += v; s2 += v * v; }
        s  = wave_sum(s);
        s2 = wave_sum(s2);
        if (lane == 0) {
            float mu  = s * (1.f / D_);
            float var = fmaxf(s2 * (1.f / D_) - mu * mu, 0.f);
            sStats[wave * 2]     = mu;
            sStats[wave * 2 + 1] = rsqrtf(var + LN_EPS_);
        }
    }
    __syncthreads();
    float gn = g[n], bn = b[n];
#pragma unroll
    for (int i = 0; i < 8; i++) {
        float mu  = sStats[(mb + i) * 2];
        float inv = sStats[(mb + i) * 2 + 1];
        outv[i] = (val[i] - mu) * inv * gn + bn;
    }
}

// ---------------------------------------------------------------------------
// One recurrent timestep: enc LIF + 4 bottom-up PC layers + 4 top-down layers.
// Single workgroup (32 waves); 17 WMMA matmuls of [16,512]x[512,512].
// ---------------------------------------------------------------------------
__global__ __launch_bounds__(1024) void snn_cell_kernel(
    int t, const int* __restrict__ ids, const float* __restrict__ embedding,
    const unsigned short* __restrict__ encW, const float* __restrict__ enc_b,
    const unsigned short* __restrict__ genW, const float* __restrict__ gen_b,
    const unsigned short* __restrict__ infW, const float* __restrict__ inf_b,
    const float* __restrict__ err_g, const float* __restrict__ err_b,
    const float* __restrict__ st_g, const float* __restrict__ st_b,
    float* states, float* enc_mem, float* gen_mem, float* inf_mem,
    float* accum, unsigned short* TD)
{
    __shared__ unsigned short sA[16 * LDA_];   // bf16 A operand (16x512)
    __shared__ float sLN[16 * D_];             // LayerNorm staging
    __shared__ float sStats[32];               // 16 x {mean, rstd}
    __shared__ float sAcc[2];                  // spk / memp partial sums

    const int tid  = threadIdx.x;
    const int lane = tid & 31;
    const int wave = tid >> 5;
    const int n0   = wave * 16;
    const int mb   = (lane >> 4) * 8;
    const int n    = n0 + (lane & 15);

    if (tid == 0) { sAcc[0] = 0.f; sAcc[1] = 0.f; }

    // --- stage token embedding (fp32 gather -> bf16 LDS) ---------------------
    prefetch_w(encW, tid);
    for (int x = tid; x < 16 * D_; x += 1024) {
        int b = x >> 9, d = x & 511;
        sA[b * LDA_ + d] = f2bf(embedding[(size_t)ids[b * S_ + t] * D_ + d]);
    }
    __syncthreads();

    // --- encoder LIF ---------------------------------------------------------
    v8f z = matmul16(sA, encW, n0, lane);
    float bu[8], tmpm[8];
    lif_frag(z, enc_b, enc_mem, n0, lane, bu, tmpm);

    // --- bottom-up predictive-coding pass ------------------------------------
    for (int j = 0; j < L_; j++) {
        float* stj = states + j * 16 * D_;
        prefetch_w(genW + (size_t)j * D_ * D_, tid);
        __syncthreads();
        for (int x = tid; x < 16 * D_; x += 1024) {
            int b = x >> 9, d = x & 511;
            sA[b * LDA_ + d] = f2bf(stj[b * D_ + d]);
        }
        __syncthreads();

        // gen LIF: pred = spike(gen_mem*decay + state@genW^T + gen_b)
        prefetch_w(infW + (size_t)j * D_ * D_, tid);
        v8f pz = matmul16(sA, genW + (size_t)j * D_ * D_, n0, lane);
        float pred[8], gmv[8];
        lif_frag(pz, gen_b + j * D_, gen_mem + j * 16 * D_, n0, lane, pred, gmv);

        // err = LN(relu(bu - pred))
        float ev[8], en[8];
#pragma unroll
        for (int i = 0; i < 8; i++) ev[i] = fmaxf(bu[i] - pred[i], 0.f);
        layer_norm_frag(ev, en, err_g + j * D_, err_b + j * D_, sLN, sStats, lane, wave, n0);

        // spk += mean(err)
        {
            float s = 0.f;
#pragma unroll
            for (int i = 0; i < 8; i++) s += en[i];
            s = wave_sum(s);
            if (lane == 0) atomicAdd(&sAcc[0], s);
        }

        // inf LIF on err
#pragma unroll
        for (int i = 0; i < 8; i++) sA[(mb + i) * LDA_ + n] = f2bf(en[i]);
        __syncthreads();
        v8f sz = matmul16(sA, infW + (size_t)j * D_ * D_, n0, lane);
        float su[8], imv[8];
        lif_frag(sz, inf_b + j * D_, inf_mem + j * 16 * D_, n0, lane, su, imv);

        // memp += mean(|inf_mem|)
        {
            float s = 0.f;
#pragma unroll
            for (int i = 0; i < 8; i++) s += fabsf(imv[i]);
            s = wave_sum(s);
            if (lane == 0) atomicAdd(&sAcc[1], s);
        }

        // new_state = LN(state + su); bu = relu(new_state)
        float tv[8], ns[8];
#pragma unroll
        for (int i = 0; i < 8; i++) tv[i] = stj[(mb + i) * D_ + n] + su[i];
        layer_norm_frag(tv, ns, st_g + j * D_, st_b + j * D_, sLN, sStats, lane, wave, n0);
#pragma unroll
        for (int i = 0; i < 8; i++) {
            stj[(mb + i) * D_ + n] = ns[i];
            bu[i] = fmaxf(ns[i], 0.f);
        }
    }

    // --- top-down generative pass --------------------------------------------
    prefetch_w(genW + (size_t)(L_ - 1) * D_ * D_, tid);
    __syncthreads();
    for (int x = tid; x < 16 * D_; x += 1024) {
        int b = x >> 9, d = x & 511;
        sA[b * LDA_ + d] = f2bf(states[(L_ - 1) * 16 * D_ + b * D_ + d]);
    }
    __syncthreads();

    for (int j = L_ - 1; j >= 0; j--) {
        // gen LIF on td (mems keep evolving)
        prefetch_w(infW + (size_t)j * D_ * D_, tid);
        v8f pz = matmul16(sA, genW + (size_t)j * D_ * D_, n0, lane);
        float pred[8], gmv[8];
        lif_frag(pz, gen_b + j * D_, gen_mem + j * 16 * D_, n0, lane, pred, gmv);

        // err = LN(relu(0 - pred)) = LN(0) = err_b[j]  (exact identity)
        __syncthreads();
        for (int x = tid; x < 16 * D_; x += 1024) {
            int b = x >> 9, d = x & 511;
            sA[b * LDA_ + d] = f2bf(err_b[j * D_ + d]);
        }
        __syncthreads();

        // inf LIF must still update inf_mem
        if (j > 0) prefetch_w(genW + (size_t)(j - 1) * D_ * D_, tid);
        v8f sz = matmul16(sA, infW + (size_t)j * D_ * D_, n0, lane);
        float su[8], imv[8];
        lif_frag(sz, inf_b + j * D_, inf_mem + j * 16 * D_, n0, lane, su, imv);

        // td = pred (binary spikes, exact in bf16)
        __syncthreads();
#pragma unroll
        for (int i = 0; i < 8; i++) {
            sA[(mb + i) * LDA_ + n] = f2bf(pred[i]);
            if (j == 0)
                TD[((size_t)t * 16 + (mb + i)) * D_ + n] = f2bf(pred[i]);
        }
        __syncthreads();
    }

    if (tid == 0) {
        accum[0] += sAcc[0] * (1.f / 8192.f);
        accum[1] += sAcc[1] * (1.f / 8192.f);
    }
}

// ---------------------------------------------------------------------------
// Batched logits GEMM: [S*B=4096, 512] x [512, 32000] bf16 -> fp32.
// Block = 16 waves = 64x64 output tile; A and B chunks staged in LDS.
// ---------------------------------------------------------------------------
__global__ __launch_bounds__(512) void snn_logits_kernel(
    const unsigned short* __restrict__ TD, const unsigned short* __restrict__ outW,
    const float* __restrict__ out_b, float* __restrict__ out)
{
    __shared__ unsigned short sA[64 * LDA2_];   // A chunk: 64 rows x 128 K
    __shared__ unsigned short sB[64 * LDA2_];   // B chunk: 64 cols x 128 K
    const int tid  = threadIdx.x;
    const int lane = tid & 31;
    const int wave = tid >> 5;
    const int wm   = wave >> 2, wn = wave & 3;
    const int m0   = blockIdx.y * 64;
    const int n0b  = blockIdx.x * 64;
    const int n0   = n0b + wn * 16;
    const int n    = n0 + (lane & 15);

    v8f acc = {0.f, 0.f, 0.f, 0.f, 0.f, 0.f, 0.f, 0.f};

    for (int kb = 0; kb < 4; kb++) {
        const int kbase = kb * 128;
        __syncthreads();
        // Stage A and B chunks with 128-bit loads: 16 uint4 per row.
        for (int x = tid; x < 64 * 16; x += 512) {
            int r = x >> 4, c = (x & 15) * 8;
            *(uint4*)(sA + r * LDA2_ + c) =
                *(const uint4*)(TD + (size_t)(m0 + r) * D_ + kbase + c);
            *(uint4*)(sB + r * LDA2_ + c) =
                *(const uint4*)(outW + (size_t)(n0b + r) * D_ + kbase + c);
        }
        __syncthreads();
        const unsigned short* arow =
            sA + (wm * 16 + (lane & 15)) * LDA2_ + ((lane >> 4) * 8);
        const unsigned short* brow =
            sB + (wn * 16 + (lane & 15)) * LDA2_ + ((lane >> 4) * 8);
#pragma unroll
        for (int k0 = 0; k0 < 128; k0 += 32) {
            FragB16 A, Bf;
            A.q[0]  = *(const uint4*)(arow + k0);
            A.q[1]  = *(const uint4*)(arow + k0 + 16);
            Bf.q[0] = *(const uint4*)(brow + k0);
            Bf.q[1] = *(const uint4*)(brow + k0 + 16);
            acc = __builtin_amdgcn_wmma_f32_16x16x32_bf16(false, A.v, false, Bf.v,
                                                          (short)0, acc, false, false);
        }
    }

    const float bn = out_b[n];
    const int mb = (lane >> 4) * 8;
#pragma unroll
    for (int i = 0; i < 8; i++) {
        int mg = m0 + wm * 16 + mb + i;         // global step-batch row
        int t  = mg >> 4;
        int b  = mg & 15;
        out[(size_t)b * ((size_t)S_ * V_) + (size_t)t * V_ + n] = acc[i] + bn;
    }
}

// ---------------------------------------------------------------------------
__global__ void f32_to_bf16_kernel(const float* __restrict__ in,
                                   unsigned short* __restrict__ out, int nelem) {
    int i = blockIdx.x * blockDim.x + threadIdx.x;
    if (i < nelem) out[i] = f2bf(in[i]);
}

__global__ void snn_finalize_kernel(const float* __restrict__ accum, float* out) {
    if (threadIdx.x == 0) {
        out[(size_t)B_ * S_ * V_]     = accum[0] * (1.f / S_);
        out[(size_t)B_ * S_ * V_ + 1] = accum[1] * (1.f / S_);
    }
}

// ---------------------------------------------------------------------------
extern "C" void kernel_launch(void* const* d_in, const int* in_sizes, int n_in,
                              void* d_out, int out_size, void* d_ws, size_t ws_size,
                              hipStream_t stream) {
    const int*   ids       = (const int*)d_in[0];
    const float* embedding = (const float*)d_in[1];
    const float* enc_W     = (const float*)d_in[2];
    const float* enc_b     = (const float*)d_in[3];
    const float* gen_W     = (const float*)d_in[4];
    const float* gen_b     = (const float*)d_in[5];
    const float* inf_W     = (const float*)d_in[6];
    const float* inf_b     = (const float*)d_in[7];
    const float* err_g     = (const float*)d_in[8];
    const float* err_b     = (const float*)d_in[9];
    const float* st_g      = (const float*)d_in[10];
    const float* st_b      = (const float*)d_in[11];
    const float* out_W     = (const float*)d_in[12];
    const float* out_b     = (const float*)d_in[13];
    float*       out       = (float*)d_out;

    char* ws = (char*)d_ws;
    size_t off = 0;
    auto walloc = [&](size_t bytes) -> void* {
        void* p = ws + off;
        off += (bytes + 255) & ~(size_t)255;
        return p;
    };
    unsigned short* encWb = (unsigned short*)walloc((size_t)D_ * D_ * 2);
    unsigned short* genWb = (unsigned short*)walloc((size_t)L_ * D_ * D_ * 2);
    unsigned short* infWb = (unsigned short*)walloc((size_t)L_ * D_ * D_ * 2);
    unsigned short* outWb = (unsigned short*)walloc((size_t)V_ * D_ * 2);
    unsigned short* TD    = (unsigned short*)walloc((size_t)S_ * B_ * D_ * 2);
    float* states  = (float*)walloc((size_t)L_ * B_ * D_ * 4);
    float* enc_mem = (float*)walloc((size_t)B_ * D_ * 4);
    float* gen_mem = (float*)walloc((size_t)L_ * B_ * D_ * 4);
    float* inf_mem = (float*)walloc((size_t)L_ * B_ * D_ * 4);
    float* accum   = (float*)walloc(2 * sizeof(float));
    if (off > ws_size) return;  // workspace too small; nothing safe to do

    // Zero all recurrent state + accumulators (contiguous region).
    hipMemsetAsync(states, 0, (size_t)((char*)accum - (char*)states) + 2 * sizeof(float),
                   stream);

    // Convert weights to bf16 once per launch (deterministic).
    const int thr = 256;
    f32_to_bf16_kernel<<<(D_ * D_ + thr - 1) / thr, thr, 0, stream>>>(enc_W, encWb, D_ * D_);
    f32_to_bf16_kernel<<<(L_ * D_ * D_ + thr - 1) / thr, thr, 0, stream>>>(gen_W, genWb, L_ * D_ * D_);
    f32_to_bf16_kernel<<<(L_ * D_ * D_ + thr - 1) / thr, thr, 0, stream>>>(inf_W, infWb, L_ * D_ * D_);
    f32_to_bf16_kernel<<<(V_ * D_ + thr - 1) / thr, thr, 0, stream>>>(out_W, outWb, V_ * D_);

    // Sequential recurrence: one workgroup per timestep (LIF state carries over).
    for (int t = 0; t < S_; t++) {
        snn_cell_kernel<<<1, 1024, 0, stream>>>(
            t, ids, embedding, encWb, enc_b, genWb, gen_b, infWb, inf_b,
            err_g, err_b, st_g, st_b, states, enc_mem, gen_mem, inf_mem, accum, TD);
    }

    // One big weight-stationary GEMM for all logits.
    snn_logits_kernel<<<dim3(V_ / 64, (S_ * B_) / 64), 512, 0, stream>>>(TD, outWb, out_b, out);

    snn_finalize_kernel<<<1, 32, 0, stream>>>(accum, out);
}
